// additive_intervention_81982335746424
// MI455X (gfx1250) — matrix operand order repaired and checked by reading, not compiled, not run on hardware
//
#include <hip/hip_runtime.h>
#include <hip/hip_bf16.h>
#include <math.h>

// Problem constants (from reference)
#define B_SZ    64
#define N_TOT   2048
#define CONDIM  512
#define FUSEDIM 1024
#define HIDDIM  512

// Main-kernel tiling: 4 waves x 2 row-tiles x 16 rows = 128 rows per block
#define ROWS    128
#define SPLITS  (N_TOT / ROWS)    // 16
#define STRIDE  520               // padded bf16 row stride in LDS (4-bank rotate/row)

typedef __attribute__((ext_vector_type(16))) __bf16 v16bf;
typedef __attribute__((ext_vector_type(8)))  float  v8f;

// Native hardware tanh (TRANS32 op on CDNA5) with compile-safe fallback.
#if __has_builtin(__builtin_amdgcn_tanhf)
#define TANHF(x) __builtin_amdgcn_tanhf(x)
#elif __has_builtin(__builtin_amdgcn_tanh_f32)
#define TANHF(x) __builtin_amdgcn_tanh_f32(x)
#else
#define TANHF(x) tanhf(x)
#endif

// B fragment: split-K layout (col-of-Wk row): chunks at +0 and +16 elems
#define LOADB128x2(dst, p)                                   \
    do {                                                     \
        ((float4*)&(dst))[0] = *(const float4*)(p);          \
        ((float4*)&(dst))[1] = *(const float4*)((p) + 16);   \
    } while (0)
// A fragment: split-K layout: chunks at +0 and +16 elems
#define LOADA128x2(dst, p)                                   \
    do {                                                     \
        ((float4*)&(dst))[0] = *(const float4*)(p);          \
        ((float4*)&(dst))[1] = *(const float4*)((p) + 16);   \
    } while (0)

// ---------------------------------------------------------------------------
// Kernel 0: Wk (f32 [HID][CON]) -> bf16 in workspace (L2-resident B operand)
// ---------------------------------------------------------------------------
__global__ __launch_bounds__(256) void cvt_wk_kernel(const float* __restrict__ Wk,
                                                     __bf16* __restrict__ wkb) {
    int i = (blockIdx.x * 256 + threadIdx.x) * 4;   // 512*512 elements total
    float4 v = *(const float4*)(Wk + i);
    __bf16* d = wkb + i;
    d[0] = (__bf16)v.x; d[1] = (__bf16)v.y; d[2] = (__bf16)v.z; d[3] = (__bf16)v.w;
}

// ---------------------------------------------------------------------------
// Kernel 1: query[b,h] = sum_f fuse_rep[b,f] * Wq[h,f]   (f32, tiny GEMM)
// ---------------------------------------------------------------------------
__global__ __launch_bounds__(256) void query_kernel(const float* __restrict__ fuse,
                                                    const float* __restrict__ Wq,
                                                    float* __restrict__ q) {
    __shared__ __align__(16) float fr[FUSEDIM];
    const int b = blockIdx.x, tid = threadIdx.x;
    for (int i = tid * 4; i < FUSEDIM; i += 256 * 4)
        *(float4*)&fr[i] = *(const float4*)(fuse + (size_t)b * FUSEDIM + i);
    __syncthreads();
    for (int h = tid; h < HIDDIM; h += 256) {
        const float* w = Wq + (size_t)h * FUSEDIM;
        float acc = 0.f;
        for (int f = 0; f < FUSEDIM; f += 4) {
            float4 wv = *(const float4*)(w + f);
            acc += wv.x * fr[f] + wv.y * fr[f + 1] + wv.z * fr[f + 2] + wv.w * fr[f + 3];
        }
        q[(size_t)b * HIDDIM + h] = acc;
    }
}

// ---------------------------------------------------------------------------
// Kernel 2 (main, fused): per (batch b, N-split blk)
//   scores = Wt . tanh(q[b] + conf_tile @ Wk^T)  via bf16 WMMA.
//   Both 16-row A tiles end up register-resident (compiler hoists the
//   loop-invariant LDS fragments); B streams from L2-resident bf16 Wk with
//   clause'd b128 loads overlapped against the matrix-pipe backlog.
//   Then block-local online-softmax partial.
// ---------------------------------------------------------------------------
__global__ __launch_bounds__(128) void fused_scores_kernel(
    const float*  __restrict__ conf,   // [B, N, CON] f32
    const float*  __restrict__ prob,   // [B, N, 1]  f32
    const float*  __restrict__ qv,     // ws: [B, HID] f32
    const __bf16* __restrict__ wkb,    // ws: [HID, CON] bf16
    const float*  __restrict__ wt,     // [HID] f32
    float* __restrict__ pacc,          // ws: [B, SPLITS, CON] f32
    float* __restrict__ pm,            // ws: [B, SPLITS] f32
    float* __restrict__ pd)            // ws: [B, SPLITS] f32
{
    __shared__ __align__(16) __bf16 confs[ROWS * STRIDE];   // ~133 KB
    __shared__ __align__(16) float  q_s[HIDDIM];
    __shared__ __align__(16) float  wt_s[HIDDIM];
    __shared__ float scores_s[ROWS];
    __shared__ float wbuf[ROWS];

    const int tid = threadIdx.x;
    const int blk = blockIdx.x;          // N split
    const int b   = blockIdx.y;          // batch
    const int n0  = blk * ROWS;
    const float* confb = conf + (size_t)b * N_TOT * CONDIM;

    // q, Wt into LDS
    for (int i = tid; i < HIDDIM; i += 128) {
        q_s[i]  = qv[(size_t)b * HIDDIM + i];
        wt_s[i] = wt[i];
    }
    // Stage conf tile (128 x 512 f32) into LDS as bf16, padded rows
    for (int g = tid; g < ROWS * CONDIM / 4; g += 128) {
        const int row = g >> 7;          // 128 float4-groups per row
        const int cg  = g & 127;
        const float4 v = *(const float4*)(confb + (size_t)(n0 + row) * CONDIM + cg * 4);
        __bf16* dst = &confs[row * STRIDE + cg * 4];
        dst[0] = (__bf16)v.x; dst[1] = (__bf16)v.y; dst[2] = (__bf16)v.z; dst[3] = (__bf16)v.w;
    }
    __syncthreads();

    const int lane = tid & 31;           // wave32
    const int wave = tid >> 5;           // 0..3, each owns 32 rows (2 tiles)
    const int half = (lane >> 4) & 1;
    const int l16  = lane & 15;
    const int r0   = wave * 32;

    float spart0[8], spart1[8];
#pragma unroll
    for (int v = 0; v < 8; ++v) { spart0[v] = 0.f; spart1[v] = 0.f; }

    // ISA 7.12.2 layouts:
    //  A 16x32 bf16: lane%16 = row; half 0 holds K 0-7,16-23; half 1 holds K 8-15,24-31
    //  B 32x16 bf16: lane%16 = col(h); half 0 holds K 0-7,16-23; half 1 holds K 8-15,24-31
    const __bf16* abase0 = &confs[(r0 + l16) * STRIDE + 8 * half];
    const __bf16* abase1 = abase0 + 16 * STRIDE;

    // Tile0 A fragments explicitly hoisted into registers.
    v16bf a0f[16];
#pragma unroll
    for (int ks = 0; ks < CONDIM / 32; ++ks)
        LOADA128x2(a0f[ks], abase0 + ks * 32);

    // Double-buffered streams with distance-2 prefetch + cross-hc prefetch.
    const __bf16* bbase = wkb + (size_t)l16 * CONDIM + 8 * half;  // hc = 0
    v16bf bb0, bb1, aa0, aa1;
    LOADB128x2(bb0, bbase);
    LOADB128x2(bb1, bbase + 32);
    LOADA128x2(aa0, abase1);
    LOADA128x2(aa1, abase1 + 32);

    for (int hc = 0; hc < HIDDIM / 16; ++hc) {
        v8f acc0 = {0.f, 0.f, 0.f, 0.f, 0.f, 0.f, 0.f, 0.f};
        v8f acc1 = {0.f, 0.f, 0.f, 0.f, 0.f, 0.f, 0.f, 0.f};
#pragma unroll
        for (int ks = 0; ks < CONDIM / 32; ks += 2) {
            acc0 = __builtin_amdgcn_wmma_f32_16x16x32_bf16(
                false, a0f[ks], false, bb0, (short)0, acc0, false, false);
            acc1 = __builtin_amdgcn_wmma_f32_16x16x32_bf16(
                false, aa0, false, bb0, (short)0, acc1, false, false);
            if (ks + 2 < CONDIM / 32) {
                LOADB128x2(bb0, bbase + (ks + 2) * 32);
                LOADA128x2(aa0, abase1 + (ks + 2) * 32);
            }
            acc0 = __builtin_amdgcn_wmma_f32_16x16x32_bf16(
                false, a0f[ks + 1], false, bb1, (short)0, acc0, false, false);
            acc1 = __builtin_amdgcn_wmma_f32_16x16x32_bf16(
                false, aa1, false, bb1, (short)0, acc1, false, false);
            if (ks + 3 < CONDIM / 32) {
                LOADB128x2(bb1, bbase + (ks + 3) * 32);
                LOADA128x2(aa1, abase1 + (ks + 3) * 32);
            }
        }
        // Prefetch next hc's first fragments; epilogue covers the latency.
        bbase += 16 * CONDIM;
        if (hc + 1 < HIDDIM / 16) {
            LOADB128x2(bb0, bbase);
            LOADB128x2(bb1, bbase + 32);
            LOADA128x2(aa0, abase1);
            LOADA128x2(aa1, abase1 + 32);
        }
        // acc[v]: row (tile base) + 8*half + v, col h = hc*16 + l16
        const float qh = q_s[hc * 16 + l16];
        const float wh = wt_s[hc * 16 + l16];
#pragma unroll
        for (int v = 0; v < 8; ++v) {
            spart0[v] += TANHF(acc0[v] + qh) * wh;
            spart1[v] += TANHF(acc1[v] + qh) * wh;
        }
    }

    // Reduce over the 16 columns held by each lane-half (masks 1,2,4,8 keep halves)
#pragma unroll
    for (int m = 1; m <= 8; m <<= 1) {
#pragma unroll
        for (int v = 0; v < 8; ++v) {
            spart0[v] += __shfl_xor(spart0[v], m, 32);
            spart1[v] += __shfl_xor(spart1[v], m, 32);
        }
    }
    if (l16 == 0) {
#pragma unroll
        for (int v = 0; v < 8; ++v) {
            scores_s[r0 + half * 8 + v]      = spart0[v];
            scores_s[r0 + 16 + half * 8 + v] = spart1[v];
        }
    }
    __syncthreads();

    // Block-local softmax partial (redundant per-thread scan of ROWS scores)
    float m_l = -3.4e38f;
    for (int r = 0; r < ROWS; ++r) m_l = fmaxf(m_l, scores_s[r]);
    float d_l = 0.f;
    for (int r = 0; r < ROWS; ++r) d_l += __expf(scores_s[r] - m_l);
    wbuf[tid] = __expf(scores_s[tid] - m_l) * prob[(size_t)b * N_TOT + n0 + tid];
    if (tid == 0) {
        pm[b * SPLITS + blk] = m_l;
        pd[b * SPLITS + blk] = d_l;
    }
    __syncthreads();

    // Weighted partial sum in f32, re-reading tile from global (L2-hot)
    const int c0 = tid * 4;              // 128 threads x 4 cols = 512
    float a0 = 0.f, a1 = 0.f, a2 = 0.f, a3 = 0.f;
    for (int r = 0; r < ROWS; ++r) {
        const float w = wbuf[r];
        const float4 cv = *(const float4*)(confb + (size_t)(n0 + r) * CONDIM + c0);
        a0 += w * cv.x; a1 += w * cv.y; a2 += w * cv.z; a3 += w * cv.w;
    }
    float* out = pacc + ((size_t)(b * SPLITS + blk)) * CONDIM + c0;
    out[0] = a0; out[1] = a1; out[2] = a2; out[3] = a3;
}

// ---------------------------------------------------------------------------
// Kernel 3: combine SPLITS partials per batch into final output
// ---------------------------------------------------------------------------
__global__ __launch_bounds__(128) void combine_kernel(const float* __restrict__ pacc,
                                                      const float* __restrict__ pm,
                                                      const float* __restrict__ pd,
                                                      float* __restrict__ outp) {
    __shared__ float scale[SPLITS];
    const int b = blockIdx.x, tid = threadIdx.x;
    float M = -3.4e38f;
    for (int l = 0; l < SPLITS; ++l) M = fmaxf(M, pm[b * SPLITS + l]);
    float D = 0.f;
    for (int l = 0; l < SPLITS; ++l) D += __expf(pm[b * SPLITS + l] - M) * pd[b * SPLITS + l];
    if (tid < SPLITS) scale[tid] = __expf(pm[b * SPLITS + tid] - M);
    __syncthreads();
    const int c0 = tid * 4;
    float a0 = 0.f, a1 = 0.f, a2 = 0.f, a3 = 0.f;
    for (int l = 0; l < SPLITS; ++l) {
        const float s = scale[l];
        const float4 v = *(const float4*)(pacc + ((size_t)(b * SPLITS + l)) * CONDIM + c0);
        a0 += s * v.x; a1 += s * v.y; a2 += s * v.z; a3 += s * v.w;
    }
    const float inv = 1.f / D;
    float* o = outp + (size_t)b * CONDIM + c0;
    o[0] = a0 * inv; o[1] = a1 * inv; o[2] = a2 * inv; o[3] = a3 * inv;
}

// ---------------------------------------------------------------------------
extern "C" void kernel_launch(void* const* d_in, const int* in_sizes, int n_in,
                              void* d_out, int out_size, void* d_ws, size_t ws_size,
                              hipStream_t stream) {
    const float* conf = (const float*)d_in[0];   // [B,N,CON]
    const float* fuse = (const float*)d_in[1];   // [B,FUSE]
    const float* prob = (const float*)d_in[2];   // [B,N,1]
    const float* Wq   = (const float*)d_in[3];   // [HID,FUSE]
    const float* Wk   = (const float*)d_in[4];   // [HID,CON]
    const float* Wt   = (const float*)d_in[5];   // [1,HID]
    float* outp = (float*)d_out;                 // [B,CON]

    // Workspace layout
    uint8_t* ws = (uint8_t*)d_ws;
    __bf16* wkb = (__bf16*)(ws);                                   // 512 KB
    float*  qv  = (float*)(ws + 512 * 1024);                       // 128 KB
    float*  pm  = (float*)(ws + 512 * 1024 + 128 * 1024);          //   8 KB
    float*  pd  = (float*)(ws + 512 * 1024 + 128 * 1024 + 8 * 1024);
    float*  pacc = (float*)(ws + 512 * 1024 + 128 * 1024 + 16 * 1024); // 2 MB
    cvt_wk_kernel<<<(HIDDIM * CONDIM / 4) / 256, 256, 0, stream>>>(Wk, wkb);
    query_kernel<<<B_SZ, 256, 0, stream>>>(fuse, Wq, qv);
    fused_scores_kernel<<<dim3(SPLITS, B_SZ), 128, 0, stream>>>(
        conf, prob, qv, wkb, Wt, pacc, pm, pd);
    combine_kernel<<<B_SZ, 128, 0, stream>>>(pacc, pm, pd, outp);
}
